// Decoder_8108898255077
// MI455X (gfx1250) — matrix operand order repaired and checked by reading, not compiled, not run on hardware
//
#include <hip/hip_runtime.h>

// Problem constants (from reference)
#define B_   16
#define T_   128
#define S_   512
#define V_   32000
#define E_   512
#define H_   1024
#define L_   2
#define G4_  4096        // 4*H
#define M_   2048        // B*T rows (row = t*B + b, time-major)

typedef __attribute__((ext_vector_type(16))) __bf16 v16bf;
typedef __attribute__((ext_vector_type(8)))  __bf16 v8bf;
typedef __attribute__((ext_vector_type(8)))  float  v8f;
typedef unsigned short ushort_t;

__device__ __forceinline__ float sigm_(float x) { return 1.0f / (1.0f + __expf(-x)); }

__device__ __forceinline__ ushort_t f2bf(float f) {
    unsigned u = __float_as_uint(f);
    u += 0x7FFFu + ((u >> 16) & 1u);   // round-to-nearest-even
    return (ushort_t)(u >> 16);
}

// ---------------------------------------------------------------- context sum
__global__ void k_ctx(const float* __restrict__ enc, float* __restrict__ ctx) {
    int idx = blockIdx.x * blockDim.x + threadIdx.x;      // B*H threads
    if (idx >= B_ * H_) return;
    int b = idx / H_, h = idx % H_;
    const float* p = enc + (size_t)b * S_ * H_ + h;
    float s = 0.f;
    for (int ss = 0; ss < S_; ++ss) s += p[(size_t)ss * H_];
    ctx[idx] = s;
}

// ------------------------------------------- f32 -> bf16 row-strided convert
__global__ void k_cvt_rows(const float* __restrict__ src, int ld, int cols,
                           ushort_t* __restrict__ dst, int n) {
    int idx = blockIdx.x * blockDim.x + threadIdx.x;
    if (idx >= n) return;
    int r = idx / cols, c = idx % cols;
    dst[idx] = f2bf(src[(size_t)r * ld + c]);
}

// ------------------------------------------------ embedding gather -> bf16 A
__global__ void k_emb(const int* __restrict__ x, const float* __restrict__ emb,
                      ushort_t* __restrict__ Aemb) {
    int idx = blockIdx.x * blockDim.x + threadIdx.x;      // M*E
    if (idx >= M_ * E_) return;
    int row = idx / E_, e = idx % E_;
    int t = row / B_, b = row % B_;
    int tok = x[b * T_ + t];
    Aemb[idx] = f2bf(emb[(size_t)tok * E_ + e]);
}

// --------------------- constA[b][j] = ctx[b] . Wih0[j, E:] + bih0[j]+bhh0[j]
__global__ void k_constA(const float* __restrict__ ctx, const float* __restrict__ Wih0,
                         const float* __restrict__ bih0, const float* __restrict__ bhh0,
                         float* __restrict__ cA) {
    int idx = blockIdx.x * blockDim.x + threadIdx.x;      // B*G4
    if (idx >= B_ * G4_) return;
    int b = idx / G4_, j = idx % G4_;
    const float* w = Wih0 + (size_t)j * (E_ + H_) + E_;
    const float* c = ctx + (size_t)b * H_;
    float s = bih0[j] + bhh0[j];
    for (int h = 0; h < H_; ++h) s += c[h] * w[h];
    cA[idx] = s;
}

// ------------------------------------------------------------ WMMA bf16 GEMM
// C[M x N] = A[M x K](bf16 rowmajor) * W^T  (W stored [N x K] bf16 rowmajor)
// Block = 256 threads = 8 waves; block tile 128(M) x 128(N); wave tile 32x64.
// K-panels (128x32 A, 128x32 B) staged to LDS via async global->LDS copies,
// double-buffered, fenced with s_wait_asynccnt + workgroup barrier.
// mode 0: C[m*N + n] (f32).  mode 1: permuted logits write out[b][t][n].
__global__ __launch_bounds__(256) void k_gemm_bf16(
        const ushort_t* __restrict__ A,
        const ushort_t* __restrict__ W,
        const float* __restrict__ bias,
        float* __restrict__ C,
        int M, int N, int K, int mode) {
    __shared__ ushort_t Asm[2][128 * 32];   // [buf][row][kk]  8KB each
    __shared__ ushort_t Bsm[2][128 * 32];

    const int tid  = threadIdx.x;
    const int lane = tid & 31;
    const int wave = tid >> 5;
    const int wm   = wave & 3;              // 4 M-strips of 32 rows
    const int wn   = wave >> 2;             // 2 N-strips of 64 cols
    const int half = lane >> 4;
    const int l16  = lane & 15;
    const int tn   = N >> 7;
    const int mblk0 = (blockIdx.x / tn) << 7;
    const int nblk0 = (blockIdx.x % tn) << 7;

    v8f acc[2][4];
    #pragma unroll
    for (int mt = 0; mt < 2; ++mt)
        #pragma unroll
        for (int nt = 0; nt < 4; ++nt) {
            float bv = bias ? bias[nblk0 + wn * 64 + nt * 16 + l16] : 0.0f;
            #pragma unroll
            for (int i = 0; i < 8; ++i) acc[mt][nt][i] = bv;
        }

    // async stage of one K-panel (A rows from mblk0, B rows from nblk0)
    auto stage = [&](int buf, int kb) {
        #pragma unroll
        for (int c = tid; c < 512; c += 256) {          // 512 chunks of 16B per tile
            int r = c >> 2, kk = (c & 3) << 3;
            {
                const ushort_t* g = A + (size_t)(mblk0 + r) * K + kb + kk;
                unsigned loff = (unsigned)(uintptr_t)(&Asm[buf][r * 32 + kk]);
                asm volatile("global_load_async_to_lds_b128 %0, %1, off"
                             :: "v"(loff), "v"((unsigned long long)(uintptr_t)g)
                             : "memory");
            }
            {
                const ushort_t* g = W + (size_t)(nblk0 + r) * K + kb + kk;
                unsigned loff = (unsigned)(uintptr_t)(&Bsm[buf][r * 32 + kk]);
                asm volatile("global_load_async_to_lds_b128 %0, %1, off"
                             :: "v"(loff), "v"((unsigned long long)(uintptr_t)g)
                             : "memory");
            }
        }
    };

    const int nk = K >> 5;
    stage(0, 0);
    asm volatile("s_wait_asynccnt 0x0" ::: "memory");
    __syncthreads();

    for (int kbi = 0; kbi < nk; ++kbi) {
        const int cur = kbi & 1;
        if (kbi + 1 < nk) stage(cur ^ 1, (kbi + 1) << 5);   // overlap copy w/ compute

        const ushort_t* As = Asm[cur];
        const ushort_t* Bs = Bsm[cur];
        v16bf afr[2];
        v16bf bfr[4];
        #pragma unroll
        for (int mt = 0; mt < 2; ++mt) {
            int r = wm * 32 + mt * 16 + l16;
            v8bf alo = *(const v8bf*)(const void*)(As + r * 32 + half * 16);
            v8bf ahi = *(const v8bf*)(const void*)(As + r * 32 + half * 16 + 8);
            #pragma unroll
            for (int i = 0; i < 8; ++i) { afr[mt][i] = alo[i]; afr[mt][i + 8] = ahi[i]; }
        }
        #pragma unroll
        for (int nt = 0; nt < 4; ++nt) {
            int r = wn * 64 + nt * 16 + l16;
            bfr[nt] = *(const v16bf*)(const void*)(Bs + r * 32 + half * 16);
        }
        #pragma unroll
        for (int mt = 0; mt < 2; ++mt)
            #pragma unroll
            for (int nt = 0; nt < 4; ++nt)
                acc[mt][nt] = __builtin_amdgcn_wmma_f32_16x16x32_bf16(
                    false, afr[mt], false, bfr[nt], (short)0, acc[mt][nt], false, false);

        asm volatile("s_wait_asynccnt 0x0" ::: "memory");
        __syncthreads();
    }

    #pragma unroll
    for (int mt = 0; mt < 2; ++mt) {
        #pragma unroll
        for (int nt = 0; nt < 4; ++nt) {
            int n = nblk0 + wn * 64 + nt * 16 + l16;
            #pragma unroll
            for (int v = 0; v < 8; ++v) {
                int m = mblk0 + wm * 32 + mt * 16 + half * 8 + v;
                if (mode == 0) {
                    C[(size_t)m * N + n] = acc[mt][nt][v];
                } else {
                    int t = m / B_, b = m % B_;
                    C[((size_t)b * T_ + t) * (size_t)V_ + n] = acc[mt][nt][v];
                }
            }
        }
    }
}

// --------------------------------------------- t=0 layer-0 step (uses G0 row)
__global__ void k_t0a(const float* __restrict__ G0, const float* __restrict__ cA,
                      const float* __restrict__ Whh0,
                      const float* __restrict__ h0in, const float* __restrict__ c0in,
                      float* __restrict__ dech0, float* __restrict__ c00,
                      ushort_t* __restrict__ H0mat, float* __restrict__ cellOut) {
    int idx = blockIdx.x * blockDim.x + threadIdx.x;      // B*H
    if (idx >= B_ * H_) return;
    int b = idx / H_, h = idx % H_;
    const float* hv = h0in + (size_t)b * H_;              // layer 0
    float s0 = 0, s1 = 0, s2 = 0, s3 = 0;
    for (int k = 0; k < H_; ++k) {
        float v = hv[k];
        s0 += v * Whh0[(size_t)(h) * H_ + k];
        s1 += v * Whh0[(size_t)(H_ + h) * H_ + k];
        s2 += v * Whh0[(size_t)(2 * H_ + h) * H_ + k];
        s3 += v * Whh0[(size_t)(3 * H_ + h) * H_ + k];
    }
    const float* g = G0 + (size_t)b * G4_;
    const float* a = cA + (size_t)b * G4_;
    float gi = g[h] + a[h] + s0;
    float gf = g[H_ + h] + a[H_ + h] + s1;
    float gg = g[2 * H_ + h] + a[2 * H_ + h] + s2;
    float go = g[3 * H_ + h] + a[3 * H_ + h] + s3;
    float c = sigm_(gf) * c0in[(size_t)b * H_ + h] + sigm_(gi) * tanhf(gg);
    float hh = sigm_(go) * tanhf(c);
    dech0[idx] = hh;
    c00[idx] = c;
    H0mat[(size_t)b * H_ + h] = f2bf(hh);                 // row (t=0)*B + b = b
    cellOut[(size_t)b * H_ + h] = c;                      // cell[0][0][b][h]
}

// ----------------- constAB[b][j] += dec_h0[b] . Whh0[j]  (in-place on constA)
__global__ void k_constB(const float* __restrict__ dech0, const float* __restrict__ Whh0,
                         float* __restrict__ cA) {
    int idx = blockIdx.x * blockDim.x + threadIdx.x;      // B*G4
    if (idx >= B_ * G4_) return;
    int b = idx / G4_, j = idx % G4_;
    const float* d = dech0 + (size_t)b * H_;
    const float* w = Whh0 + (size_t)j * H_;
    float s = 0.f;
    for (int k = 0; k < H_; ++k) s += d[k] * w[k];
    cA[idx] += s;
}

// --------------------------------------------- t=0 layer-1 step (scalar f32)
__global__ void k_t0b(const float* __restrict__ dech0,
                      const float* __restrict__ Wih1, const float* __restrict__ Whh1,
                      const float* __restrict__ bih1, const float* __restrict__ bhh1,
                      const float* __restrict__ h0in, const float* __restrict__ c0in,
                      float* __restrict__ dech1, float* __restrict__ c10,
                      ushort_t* __restrict__ H1mat, float* __restrict__ cellOut) {
    int idx = blockIdx.x * blockDim.x + threadIdx.x;      // B*H
    if (idx >= B_ * H_) return;
    int b = idx / H_, h = idx % H_;
    const float* xin = dech0 + (size_t)b * H_;
    const float* hv  = h0in + (size_t)(B_ + b) * H_;      // layer 1 of h0
    float s[4] = {0, 0, 0, 0};
    for (int q = 0; q < 4; ++q) {
        int j = q * H_ + h;
        const float* wi = Wih1 + (size_t)j * H_;
        const float* wh = Whh1 + (size_t)j * H_;
        float acc = bih1[j] + bhh1[j];
        for (int k = 0; k < H_; ++k) acc += xin[k] * wi[k] + hv[k] * wh[k];
        s[q] = acc;
    }
    float c = sigm_(s[1]) * c0in[(size_t)(B_ + b) * H_ + h] + sigm_(s[0]) * tanhf(s[2]);
    float hh = sigm_(s[3]) * tanhf(c);
    dech1[idx] = hh;
    c10[idx] = c;
    H1mat[(size_t)b * H_ + h] = f2bf(hh);
    cellOut[((size_t)T_ * B_ + b) * H_ + h] = c;          // cell[1][0][b][h]
}

// --------------- constC[b][j] = dec_h1[b] . Whh1[j] + bih1[j] + bhh1[j]
__global__ void k_constC(const float* __restrict__ dech1, const float* __restrict__ Whh1,
                         const float* __restrict__ bih1, const float* __restrict__ bhh1,
                         float* __restrict__ cC) {
    int idx = blockIdx.x * blockDim.x + threadIdx.x;      // B*G4
    if (idx >= B_ * G4_) return;
    int b = idx / G4_, j = idx % G4_;
    const float* d = dech1 + (size_t)b * H_;
    const float* w = Whh1 + (size_t)j * H_;
    float s = bih1[j] + bhh1[j];
    for (int k = 0; k < H_; ++k) s += d[k] * w[k];
    cC[idx] = s;
}

// ------------------------- elementwise cell scan over t (layer selects outs)
__global__ void k_scan(const float* __restrict__ G, const float* __restrict__ cst,
                       const float* __restrict__ cInit,
                       ushort_t* __restrict__ Hmat, float* __restrict__ cellOut,
                       int layer) {
    int idx = blockIdx.x * blockDim.x + threadIdx.x;      // B*H
    if (idx >= B_ * H_) return;
    int b = idx / H_, h = idx % H_;
    const float* a = cst + (size_t)b * G4_;
    float ai = a[h], af = a[H_ + h], ag = a[2 * H_ + h], ao = a[3 * H_ + h];
    float c = cInit[idx];
    float* cellBase = cellOut + (size_t)layer * T_ * B_ * H_ + (size_t)b * H_ + h;
    for (int t = 1; t < T_; ++t) {
        size_t row = (size_t)t * B_ + b;
        const float* g = G + row * G4_;
        float gi = g[h] + ai;
        float gf = g[H_ + h] + af;
        float gg = g[2 * H_ + h] + ag;
        float go = g[3 * H_ + h] + ao;
        c = sigm_(gf) * c + sigm_(gi) * tanhf(gg);
        float hh = sigm_(go) * tanhf(c);
        Hmat[row * H_ + h] = f2bf(hh);
        cellBase[(size_t)t * B_ * H_] = c;
    }
}

// ------------------------------------------ in-place log_softmax over V rows
__global__ void k_logsoftmax(float* __restrict__ outp) {
    __shared__ float red[256];
    int tid = threadIdx.x;
    float* p = outp + (size_t)blockIdx.x * V_;
    float m = -3.402823466e38f;
    for (int i = tid; i < V_; i += 256) m = fmaxf(m, p[i]);
    red[tid] = m; __syncthreads();
    for (int s = 128; s > 0; s >>= 1) { if (tid < s) red[tid] = fmaxf(red[tid], red[tid + s]); __syncthreads(); }
    m = red[0]; __syncthreads();
    float sum = 0.f;
    for (int i = tid; i < V_; i += 256) sum += __expf(p[i] - m);
    red[tid] = sum; __syncthreads();
    for (int s = 128; s > 0; s >>= 1) { if (tid < s) red[tid] += red[tid + s]; __syncthreads(); }
    float lse = m + __logf(red[0]);
    __syncthreads();
    for (int i = tid; i < V_; i += 256) p[i] -= lse;
}

// --------------- hidden_states (L, T+1, B, H): [l][0]=h0, [l][1..T]=dec_h[l]
__global__ void k_states(const float* __restrict__ h0in,
                         const float* __restrict__ dech0, const float* __restrict__ dech1,
                         float* __restrict__ hidOut) {
    const int N = L_ * (T_ + 1) * B_ * H_;
    int idx = blockIdx.x * blockDim.x + threadIdx.x;
    if (idx >= N) return;
    int h = idx % H_;
    int b = (idx / H_) % B_;
    int tt = (idx / (H_ * B_)) % (T_ + 1);
    int l = idx / (H_ * B_ * (T_ + 1));
    float v;
    if (tt == 0)      v = h0in[(size_t)(l * B_ + b) * H_ + h];
    else if (l == 0)  v = dech0[(size_t)b * H_ + h];
    else              v = dech1[(size_t)b * H_ + h];
    hidOut[idx] = v;
}

// ============================================================================
extern "C" void kernel_launch(void* const* d_in, const int* in_sizes, int n_in,
                              void* d_out, int out_size, void* d_ws, size_t ws_size,
                              hipStream_t stream) {
    (void)in_sizes; (void)n_in; (void)out_size; (void)ws_size;
    const int*   x     = (const int*)d_in[0];
    const float* enc   = (const float*)d_in[2];
    const float* h0in  = (const float*)d_in[3];
    const float* c0in  = (const float*)d_in[4];
    const float* emb   = (const float*)d_in[5];
    const float* Wih0  = (const float*)d_in[6];
    const float* Whh0  = (const float*)d_in[7];
    const float* bih0  = (const float*)d_in[8];
    const float* bhh0  = (const float*)d_in[9];
    const float* Wih1  = (const float*)d_in[10];
    const float* Whh1  = (const float*)d_in[11];
    const float* bih1  = (const float*)d_in[12];
    const float* bhh1  = (const float*)d_in[13];
    const float* fcW   = (const float*)d_in[14];
    const float* fcb   = (const float*)d_in[15];

    float* out       = (float*)d_out;
    float* outLogits = out;                                        // (B,T,V)
    float* outHidden = out + (size_t)B_ * T_ * V_;                 // (L,T+1,B,H)
    float* outCell   = outHidden + (size_t)L_ * (T_ + 1) * B_ * H_;// (L,T,B,H)

    // ---- workspace carve-up (256B aligned) -------------------------------
    char* wsb = (char*)d_ws;
    size_t off = 0;
    auto carve = [&](size_t bytes) { char* p = wsb + off; off += (bytes + 255) & ~(size_t)255; return p; };
    float*    ctx    = (float*)carve((size_t)B_ * H_ * 4);
    float*    cA     = (float*)carve((size_t)B_ * G4_ * 4);   // constA -> constAB
    float*    cC     = (float*)carve((size_t)B_ * G4_ * 4);
    float*    dech0  = (float*)carve((size_t)B_ * H_ * 4);
    float*    dech1  = (float*)carve((size_t)B_ * H_ * 4);
    float*    c00    = (float*)carve((size_t)B_ * H_ * 4);
    float*    c10    = (float*)carve((size_t)B_ * H_ * 4);
    ushort_t* WEb    = (ushort_t*)carve((size_t)G4_ * E_ * 2);
    ushort_t* Wih1b  = (ushort_t*)carve((size_t)G4_ * H_ * 2);
    ushort_t* fcWb   = (ushort_t*)carve((size_t)V_ * H_ * 2);
    ushort_t* Aemb   = (ushort_t*)carve((size_t)M_ * E_ * 2);
    ushort_t* H0mat  = (ushort_t*)carve((size_t)M_ * H_ * 2);
    ushort_t* H1mat  = (ushort_t*)carve((size_t)M_ * H_ * 2);
    float*    G      = (float*)carve((size_t)M_ * G4_ * 4);   // reused G0 then G1

    const int TPB = 256;
    auto blk = [](long long n) { return (unsigned)((n + 255) / 256); };

    // 1) context
    k_ctx<<<blk(B_ * H_), TPB, 0, stream>>>(enc, ctx);
    // 2) weight/activation conversions to bf16
    k_cvt_rows<<<blk((long long)G4_ * E_), TPB, 0, stream>>>(Wih0, E_ + H_, E_, WEb, G4_ * E_);
    k_cvt_rows<<<blk((long long)G4_ * H_), TPB, 0, stream>>>(Wih1, H_, H_, Wih1b, G4_ * H_);
    k_cvt_rows<<<blk((long long)V_ * H_), TPB, 0, stream>>>(fcW, H_, H_, fcWb, V_ * H_);
    k_emb<<<blk((long long)M_ * E_), TPB, 0, stream>>>(x, emb, Aemb);
    // 3) constA = ctx-part + biases
    k_constA<<<blk(B_ * G4_), TPB, 0, stream>>>(ctx, Wih0, bih0, bhh0, cA);
    // 4) G0 = Aemb @ WE^T   (WMMA+async-LDS, M=2048 N=4096 K=512)
    k_gemm_bf16<<<(M_ / 128) * (G4_ / 128), TPB, 0, stream>>>(Aemb, WEb, nullptr, G, M_, G4_, E_, 0);
    // 5) t=0 layer-0
    k_t0a<<<blk(B_ * H_), TPB, 0, stream>>>(G, cA, Whh0, h0in, c0in, dech0, c00, H0mat, outCell);
    // 6) constAB += dec_h0 . Whh0
    k_constB<<<blk(B_ * G4_), TPB, 0, stream>>>(dech0, Whh0, cA);
    // 7) layer-0 cell scan t=1..T-1 -> H0mat, cell[0]
    k_scan<<<blk(B_ * H_), TPB, 0, stream>>>(G, cA, c00, H0mat, outCell, 0);
    // 8) t=0 layer-1 (scalar)
    k_t0b<<<blk(B_ * H_), TPB, 0, stream>>>(dech0, Wih1, Whh1, bih1, bhh1, h0in, c0in,
                                            dech1, c10, H1mat, outCell);
    // 9) G1 = H0mat @ Wih1^T  (WMMA, K=1024; reuses G after scan0 consumed it)
    k_gemm_bf16<<<(M_ / 128) * (G4_ / 128), TPB, 0, stream>>>(H0mat, Wih1b, nullptr, G, M_, G4_, H_, 0);
    // 10) constC
    k_constC<<<blk(B_ * G4_), TPB, 0, stream>>>(dech1, Whh1, bih1, bhh1, cC);
    // 11) layer-1 cell scan -> H1mat, cell[1]
    k_scan<<<blk(B_ * H_), TPB, 0, stream>>>(G, cC, c10, H1mat, outCell, 1);
    // 12) logits = H1 @ fcW^T + fcb, written permuted into (B,T,V)  (WMMA, big)
    k_gemm_bf16<<<(M_ / 128) * (V_ / 128), TPB, 0, stream>>>(H1mat, fcWb, fcb, outLogits, M_, V_, H_, 1);
    // 13) in-place log_softmax over each (b,t) row
    k_logsoftmax<<<B_ * T_, TPB, 0, stream>>>(outLogits);
    // 14) hidden_states
    k_states<<<blk((long long)L_ * (T_ + 1) * B_ * H_), TPB, 0, stream>>>(h0in, dech0, dech1, outHidden);
}
